// CustomConv1d_82085414961669
// MI455X (gfx1250) — matrix operand order repaired and checked by reading, not compiled, not run on hardware
//
#include <hip/hip_runtime.h>

// ---------------------------------------------------------------------------
// Conv1d (B=16, C_IN=256, L=4096, C_OUT=256, K=7, pad=3) as implicit GEMM
// on gfx1250 using v_wmma_f32_16x16x32_bf16 with bf16x3 split-precision
// (hi/lo decomposition) so the result tracks the f32 reference to ~1e-5.
//
// GEMM view: out[(b,l), co] = sum_{k=0..6} sum_{ci} x[b,ci,l+k-3] * w[co,ci,k]
//  -> 7 shifted GEMMs with K-dim = C_IN = 256 (8 WMMA K-steps of 32 each).
//
// Roofline: 60 GFLOP, 134 MB minimal traffic (5.8 us @ 23.3 TB/s); problem
// fits in 192 MB L2, so bf16 matrix throughput is the binding resource.
// Staging latency for the next x chunk is hidden with global_prefetch_b8
// issued at the top of the 7-tap compute section.
// ---------------------------------------------------------------------------

typedef __attribute__((ext_vector_type(8)))  __bf16 v8bf;
typedef __attribute__((ext_vector_type(16))) __bf16 v16bf;
typedef __attribute__((ext_vector_type(8)))  float  v8f;

#define B_      16
#define C_IN_   256
#define L_      4096
#define C_OUT_  256
#define KW_     7
#define PAD_    3

#define BM 128          // output l-tile per block
#define BN 128          // output co-tile per block
#define AP 40           // LDS row pitch (ushorts) for x tile: 80 B, 16B-aligned
#define AROWS 136       // 128 + 6 halo rows (rounded up)

__device__ __forceinline__ unsigned short f32_bf16(float f) {
  unsigned int u = __float_as_uint(f);
  u += 0x7FFFu + ((u >> 16) & 1u);          // round-to-nearest-even
  return (unsigned short)(u >> 16);
}
__device__ __forceinline__ float bf16_f32(unsigned short h) {
  return __uint_as_float(((unsigned int)h) << 16);
}
__device__ __forceinline__ v16bf cat8(v8bf a, v8bf b) {
  v16bf r;
#pragma unroll
  for (int i = 0; i < 8; ++i) { r[i] = a[i]; r[i + 8] = b[i]; }
  return r;
}

// One-time weight prepack: f32 w[co][ci][k]  ->  bf16 hi/lo w[k][co][ci]
// so B fragments are contiguous 32-B v16bf loads per lane.
__global__ void conv1d_prepack_w(const float* __restrict__ w,
                                 unsigned short* __restrict__ whi,
                                 unsigned short* __restrict__ wlo) {
  int idx = blockIdx.x * 256 + threadIdx.x;   // 7*256*256 = 458752 threads
  int ci = idx & 255;
  int co = (idx >> 8) & 255;
  int k  = idx >> 16;
  float v = w[((size_t)co * C_IN_ + ci) * KW_ + k];
  unsigned short h = f32_bf16(v);
  unsigned short l = f32_bf16(v - bf16_f32(h));
  size_t d = ((size_t)k * C_OUT_ + co) * C_IN_ + ci;
  whi[d] = h;
  wlo[d] = l;
}

__global__ __launch_bounds__(256)
void conv1d_wmma_bf16x3(const float* __restrict__ x,
                        const unsigned short* __restrict__ whi,
                        const unsigned short* __restrict__ wlo,
                        const float* __restrict__ bias,
                        float* __restrict__ out) {
  // x tile staged transposed to [l_local][ci_local], split hi/lo bf16.
  __shared__ unsigned short a_hi[AROWS * AP];   // 10,880 B
  __shared__ unsigned short a_lo[AROWS * AP];   // 10,880 B

  const int t    = threadIdx.x;
  const int lane = t & 31;
  const int wave = t >> 5;      // 8 waves
  const int wm   = wave & 3;    // wave M index: 4 x 32 rows
  const int wn   = wave >> 2;   // wave N index: 2 x 64 cols
  const int r    = lane & 15;
  const int sel  = lane >> 4;   // half-wave select

  const int l0 = blockIdx.x * BM;
  const int n0 = blockIdx.y * BN;
  const int b  = blockIdx.z;

  v8f acc[2][4];
#pragma unroll
  for (int i = 0; i < 2; ++i)
#pragma unroll
    for (int j = 0; j < 4; ++j)
#pragma unroll
      for (int e = 0; e < 8; ++e) acc[i][j][e] = 0.0f;

  const int ci_ld = t >> 3;     // 0..31: which ci row this thread stages
  const int j0    = t & 7;      // position within the 8-thread row group

  for (int ci0 = 0; ci0 < C_IN_; ci0 += 32) {
    __syncthreads();
    {
      const float* xrow = x + ((size_t)b * C_IN_ + (ci0 + ci_ld)) * L_;
      for (int jl = j0; jl < BM + 2 * PAD_; jl += 8) {
        int l = l0 - PAD_ + jl;
        float v = (l >= 0 && l < L_) ? xrow[l] : 0.0f;
        unsigned short h  = f32_bf16(v);
        unsigned short lo = f32_bf16(v - bf16_f32(h));
        a_hi[jl * AP + ci_ld] = h;
        a_lo[jl * AP + ci_ld] = lo;
      }
    }
    __syncthreads();

    // Pull the next ci-chunk's x rows toward L1/L2 while the 7-tap WMMA
    // section below runs: 8 prefetch points per row at 128-B spacing cover
    // the 536-B halo row; 256 threads cover the whole 32-row tile.
    if (ci0 + 32 < C_IN_) {
      const float* xnext = x + ((size_t)b * C_IN_ + (ci0 + 32 + ci_ld)) * L_;
      int lp = l0 - PAD_ + j0 * 32;
      lp = lp < 0 ? 0 : (lp > L_ - 1 ? L_ - 1 : lp);
      __builtin_prefetch(xnext + lp, 0, 0);
    }

    for (int k = 0; k < KW_; ++k) {
      // A fragments (16x32 bf16): lane<16 holds K{0..15}, lane>=16
      // holds K{16..31}; row = m + (tap shift k). Two ds_load_b128 each.
      v16bf Ah[2], Al[2];
#pragma unroll
      for (int tm = 0; tm < 2; ++tm) {
        int off = (wm * 32 + tm * 16 + k + r) * AP + sel * 16;
        Ah[tm] = cat8(*(const v8bf*)&a_hi[off], *(const v8bf*)&a_hi[off + 8]);
        Al[tm] = cat8(*(const v8bf*)&a_lo[off], *(const v8bf*)&a_lo[off + 8]);
      }
      // B fragments (32x16 bf16): lane = column co; lanes 0-15 hold K 0..15,
      // lanes 16-31 hold K 16..31 -> one contiguous 32-B v16bf load per lane.
      v16bf Bh[4], Bl[4];
#pragma unroll
      for (int tn = 0; tn < 4; ++tn) {
        size_t wr = ((size_t)(k * C_OUT_ + n0 + wn * 64 + tn * 16 + r)) * C_IN_
                    + ci0 + sel * 16;
        Bh[tn] = *(const v16bf*)(whi + wr);
        Bl[tn] = *(const v16bf*)(wlo + wr);
      }
      // bf16x3: hi*hi + hi*lo + lo*hi, f32 accumulate.
#pragma unroll
      for (int tm = 0; tm < 2; ++tm)
#pragma unroll
        for (int tn = 0; tn < 4; ++tn) {
          acc[tm][tn] = __builtin_amdgcn_wmma_f32_16x16x32_bf16(
              false, Ah[tm], false, Bh[tn], (short)0, acc[tm][tn], false, false);
          acc[tm][tn] = __builtin_amdgcn_wmma_f32_16x16x32_bf16(
              false, Ah[tm], false, Bl[tn], (short)0, acc[tm][tn], false, false);
          acc[tm][tn] = __builtin_amdgcn_wmma_f32_16x16x32_bf16(
              false, Al[tm], false, Bh[tn], (short)0, acc[tm][tn], false, false);
        }
    }
  }

  // Epilogue: C/D layout -> element e of v8f is row M = e + 8*sel, col = lane&15.
  // Rows map to consecutive l, so each lane stores two float4s per tile.
#pragma unroll
  for (int tm = 0; tm < 2; ++tm)
#pragma unroll
    for (int tn = 0; tn < 4; ++tn) {
      int co = n0 + wn * 64 + tn * 16 + r;
      float bv = bias[co];
      int lb = l0 + wm * 32 + tm * 16 + 8 * sel;
      float* op = out + ((size_t)b * C_OUT_ + co) * L_ + lb;
      v8f c = acc[tm][tn];
      float4 v0 = make_float4(c[0] + bv, c[1] + bv, c[2] + bv, c[3] + bv);
      float4 v1 = make_float4(c[4] + bv, c[5] + bv, c[6] + bv, c[7] + bv);
      *(float4*)(op)     = v0;
      *(float4*)(op + 4) = v1;
    }
}

extern "C" void kernel_launch(void* const* d_in, const int* in_sizes, int n_in,
                              void* d_out, int out_size, void* d_ws, size_t ws_size,
                              hipStream_t stream) {
  (void)in_sizes; (void)n_in; (void)out_size; (void)ws_size;
  const float* x    = (const float*)d_in[0];
  const float* w    = (const float*)d_in[1];
  const float* bias = (const float*)d_in[2];

  unsigned short* whi = (unsigned short*)d_ws;                    // 917,504 B
  unsigned short* wlo = whi + (size_t)KW_ * C_OUT_ * C_IN_;       // +917,504 B

  conv1d_prepack_w<<<(KW_ * C_OUT_ * C_IN_) / 256, 256, 0, stream>>>(w, whi, wlo);

  dim3 grid(L_ / BM, C_OUT_ / BN, B_);   // 32 x 2 x 16 = 1024 blocks
  conv1d_wmma_bf16x3<<<grid, 256, 0, stream>>>(x, whi, wlo, bias, (float*)d_out);
}
